// HyperbolicGRUCell_43095701848671
// MI455X (gfx1250) — matrix operand order repaired and checked by reading, not compiled, not run on hardware
//
#include <hip/hip_runtime.h>
#include <hip/hip_bf16.h>
#include <math.h>

typedef __attribute__((ext_vector_type(16))) _Float16 v16h;
typedef __attribute__((ext_vector_type(8)))  _Float16 v8h;
typedef __attribute__((ext_vector_type(8)))  float    v8f;

// builtin_amdgcn_global_load_async_to_lds_b128 takes pointers to 16B int vectors
typedef int v4i_ __attribute__((vector_size(16)));
typedef __attribute__((address_space(1))) v4i_ gv4i;   // global
typedef __attribute__((address_space(3))) v4i_ lv4i;   // LDS

#define B_ROWS 32768
#define HDIM   512
#define GDIM   1536   // 3*H
#define SQRT_C 0.1f   // sqrt(0.01)
#define M_BLK  64     // rows of h_new per block

// LDS row stride for A tiles: 520 halves = 1040 B = 260 dwords.
// 260 % 64 = 4 -> each row rotates 4 banks; 16 fragment rows cover all 64 banks.
#define LDS_ROW 520
#define LDS_H_OFF (M_BLK * LDS_ROW)   // h-tile base (in halves)

// ---------------------------------------------------------------------------
// CDNA5 async-copy helpers (ASYNCcnt path), with safe fallbacks
// ---------------------------------------------------------------------------
__device__ __forceinline__ void async_cp16(void* l, const void* g) {
#if __has_builtin(__builtin_amdgcn_global_load_async_to_lds_b128)
    __builtin_amdgcn_global_load_async_to_lds_b128(
        (gv4i*)g, (lv4i*)l, /*offset=*/0, /*cpol=*/0);
#else
    *(int4*)l = *(const int4*)g;   // synchronous fallback
#endif
}

__device__ __forceinline__ void wait_async_all() {
#if __has_builtin(__builtin_amdgcn_s_wait_asynccnt)
    __builtin_amdgcn_s_wait_asynccnt(0);
#else
    asm volatile("s_wait_asynccnt 0" ::: "memory");
#endif
}

// ---------------------------------------------------------------------------
// Fast transcendentals (single-instruction V_EXP_F32 / V_LOG_F32 TRANS path;
// TRANS ops co-execute with WMMA per the CDNA5 scheduling model)
// ---------------------------------------------------------------------------
__device__ __forceinline__ float fast_sigmoid(float v) {
    return 1.0f / (1.0f + __expf(-v));
}
__device__ __forceinline__ float fast_tanh(float v) {
    float e = __expf(-2.0f * v);
    return (1.0f - e) / (1.0f + e);
}

// ---------------------------------------------------------------------------
// Kernel 0: convert weights fp32 -> f16 (3 MB total; lives in L2 afterwards)
// ---------------------------------------------------------------------------
__global__ __launch_bounds__(256) void cvt_weights_kernel(
    const float* __restrict__ wih, const float* __restrict__ whh,
    _Float16* __restrict__ wih16, _Float16* __restrict__ whh16)
{
    const int NW = GDIM * HDIM;
    int idx = blockIdx.x * 256 + threadIdx.x;
    if (idx < NW) {
        wih16[idx] = (_Float16)wih[idx];
        whh16[idx] = (_Float16)whh[idx];
    }
}

// ---------------------------------------------------------------------------
// Kernel 1: per-row log_map_zero for x and h.
// ---------------------------------------------------------------------------
__device__ __forceinline__ float log_scale(float nrm) {
    float nm = fmaxf(nrm, 1e-15f);
    float t  = SQRT_C * nm;
    t = fminf(t, 1.0f - 1e-5f);
    // artanh(t) = 0.5 * log((1+t)/(1-t)) ; one V_LOG_F32
    float art = 0.5f * __logf((1.0f + t) / (1.0f - t));
    return art / (SQRT_C * nm);
}

__global__ __launch_bounds__(256) void row_log_map_kernel(
    const float* __restrict__ x, const float* __restrict__ h,
    _Float16* __restrict__ xt16, _Float16* __restrict__ ht16,
    float* __restrict__ htf32)
{
    __shared__ float red[256];
    const int row = blockIdx.x;
    const int t   = threadIdx.x;
    const size_t base = (size_t)row * HDIM;

    float x0 = x[base + t], x1 = x[base + t + 256];
    float h0 = h[base + t], h1 = h[base + t + 256];

    red[t] = x0 * x0 + x1 * x1;
    __syncthreads();
    for (int s = 128; s > 0; s >>= 1) {
        if (t < s) red[t] += red[t + s];
        __syncthreads();
    }
    float nx = sqrtf(red[0]);
    __syncthreads();

    red[t] = h0 * h0 + h1 * h1;
    __syncthreads();
    for (int s = 128; s > 0; s >>= 1) {
        if (t < s) red[t] += red[t + s];
        __syncthreads();
    }
    float nh = sqrtf(red[0]);

    float sx = log_scale(nx);
    float sh = log_scale(nh);

    xt16[base + t]        = (_Float16)(sx * x0);
    xt16[base + t + 256]  = (_Float16)(sx * x1);
    ht16[base + t]        = (_Float16)(sh * h0);
    ht16[base + t + 256]  = (_Float16)(sh * h1);
    htf32[base + t]       = sh * h0;
    htf32[base + t + 256] = sh * h1;
}

// ---------------------------------------------------------------------------
// Kernel 2: fused dual GEMM (x_t@Wih^T, h_t@Whh^T) + GRU gating.
// Block = 256 threads = 8 waves; block tile = 64 rows x 64 cols of h_new.
// Each wave owns TWO 16x16 m-tiles against one 16-col B slice, so every
// B fragment load feeds two WMMAs (12 WMMA per k-slab per wave).
// A tiles (x_t, h_t; 64 rows x K=512, 130 KB) staged once per block into LDS
// via GLOBAL_LOAD_ASYNC_TO_LDS_B128 (ASYNCcnt); fragments read via ds_load.
// B tiles (weights, 3 MB) stream from the 192 MB L2.
// ---------------------------------------------------------------------------
__device__ __forceinline__ v16h load_a_frag(const _Float16* p) {
    union { v16h v; v8h h[2]; } u;
    u.h[0] = *(const v8h*)(p);
    u.h[1] = *(const v8h*)(p + 8);
    return u.v;
}

__device__ __forceinline__ v8f wmma16(v16h a, v16h b, v8f c) {
    return __builtin_amdgcn_wmma_f32_16x16x32_f16(
        false, a, false, b, (short)0, c, false, false);
}

__global__ __launch_bounds__(256) void gru_gemm_kernel(
    const _Float16* __restrict__ xt, const _Float16* __restrict__ ht,
    const _Float16* __restrict__ wih, const _Float16* __restrict__ whh,
    const float* __restrict__ bih, const float* __restrict__ bhh,
    const float* __restrict__ htf, float* __restrict__ hnew)
{
    __shared__ _Float16 lds_a[2 * M_BLK * LDS_ROW];   // x tile | h tile (130 KB)

    const int wave = threadIdx.x >> 5;
    const int lane = threadIdx.x & 31;
    const int kh   = lane >> 4;
    const int lrel = lane & 15;

    const int m_blk  = blockIdx.y * M_BLK;
    const int n_base = blockIdx.x * 64 + (wave & 3) * 16;
    const int m_grp  = (wave >> 2) * 32;      // this wave's 32-row group

    // ---- async stage: 64 contiguous rows of x_t and h_t -> LDS -------------
    {
        const char* gx = (const char*)(xt + (size_t)m_blk * HDIM);
        const char* gh = (const char*)(ht + (size_t)m_blk * HDIM);
        char* lx = (char*)&lds_a[0];
        char* lh = (char*)&lds_a[LDS_H_OFF];
        #pragma unroll
        for (int i = 0; i < 16; ++i) {
            int c   = threadIdx.x + 256 * i;   // 0..4095 16B-chunks per matrix
            int row = c >> 6;                  // 0..63
            int col = (c & 63) << 4;           // byte offset within row
            async_cp16(lx + row * (LDS_ROW * 2) + col, gx + row * 1024 + col);
            async_cp16(lh + row * (LDS_ROW * 2) + col, gh + row * 1024 + col);
        }
        wait_async_all();
        __syncthreads();
    }

    // A fragment base pointers in LDS: rows r0 = m_grp+lrel, r1 = r0+16
    const _Float16* lax0 = &lds_a[0]         + (size_t)(m_grp + lrel)      * LDS_ROW + kh * 16;
    const _Float16* lax1 = &lds_a[0]         + (size_t)(m_grp + 16 + lrel) * LDS_ROW + kh * 16;
    const _Float16* lah0 = &lds_a[LDS_H_OFF] + (size_t)(m_grp + lrel)      * LDS_ROW + kh * 16;
    const _Float16* lah1 = &lds_a[LDS_H_OFF] + (size_t)(m_grp + 16 + lrel) * LDS_ROW + kh * 16;

    // B operands: row n of W (row-major, K contiguous); K slab kh*16..kh*16+15
    const _Float16* wr = wih + (size_t)(n_base        + lrel) * HDIM + kh * 16;
    const _Float16* wz = wih + (size_t)(n_base +  512 + lrel) * HDIM + kh * 16;
    const _Float16* wn = wih + (size_t)(n_base + 1024 + lrel) * HDIM + kh * 16;
    const _Float16* ur = whh + (size_t)(n_base        + lrel) * HDIM + kh * 16;
    const _Float16* uz = whh + (size_t)(n_base +  512 + lrel) * HDIM + kh * 16;
    const _Float16* un = whh + (size_t)(n_base + 1024 + lrel) * HDIM + kh * 16;

    v8f axr0 = {}, axz0 = {}, axn0 = {}, ahr0 = {}, ahz0 = {}, ahn0 = {};
    v8f axr1 = {}, axz1 = {}, axn1 = {}, ahr1 = {}, ahz1 = {}, ahn1 = {};

    #pragma unroll 2
    for (int kb = 0; kb < HDIM; kb += 32) {
        v16h ax0 = load_a_frag(lax0 + kb);
        v16h ax1 = load_a_frag(lax1 + kb);
        v16h ah0 = load_a_frag(lah0 + kb);
        v16h ah1 = load_a_frag(lah1 + kb);

        v16h br = *(const v16h*)(wr + kb);
        axr0 = wmma16(ax0, br, axr0);
        axr1 = wmma16(ax1, br, axr1);
        v16h bz = *(const v16h*)(wz + kb);
        axz0 = wmma16(ax0, bz, axz0);
        axz1 = wmma16(ax1, bz, axz1);
        v16h bn = *(const v16h*)(wn + kb);
        axn0 = wmma16(ax0, bn, axn0);
        axn1 = wmma16(ax1, bn, axn1);
        v16h cr = *(const v16h*)(ur + kb);
        ahr0 = wmma16(ah0, cr, ahr0);
        ahr1 = wmma16(ah1, cr, ahr1);
        v16h cz = *(const v16h*)(uz + kb);
        ahz0 = wmma16(ah0, cz, ahz0);
        ahz1 = wmma16(ah1, cz, ahz1);
        v16h cn = *(const v16h*)(un + kb);
        ahn0 = wmma16(ah0, cn, ahn0);
        ahn1 = wmma16(ah1, cn, ahn1);
    }

    // Epilogue. C/D layout: VGPR v -> M = v + 8*kh, N = lrel.
    const int n = n_base + lrel;
    const float bir = bih[n], biz = bih[n + 512], bin_ = bih[n + 1024];
    const float bhr = bhh[n], bhz = bhh[n + 512], bhn = bhh[n + 1024];
    const int m0 = m_blk + m_grp + kh * 8;          // first m-tile
    const int m1 = m0 + 16;                         // second m-tile

    #pragma unroll
    for (int v = 0; v < 8; ++v) {
        {
            const int m = m0 + v;
            float r  = fast_sigmoid((axr0[v] + bir) + (ahr0[v] + bhr));
            float z  = fast_sigmoid((axz0[v] + biz) + (ahz0[v] + bhz));
            float nn = fast_tanh((axn0[v] + bin_) + r * (ahn0[v] + bhn));
            float h0 = htf[(size_t)m * HDIM + n];
            hnew[(size_t)m * HDIM + n] = (1.0f - z) * nn + z * h0;
        }
        {
            const int m = m1 + v;
            float r  = fast_sigmoid((axr1[v] + bir) + (ahr1[v] + bhr));
            float z  = fast_sigmoid((axz1[v] + biz) + (ahz1[v] + bhz));
            float nn = fast_tanh((axn1[v] + bin_) + r * (ahn1[v] + bhn));
            float h0 = htf[(size_t)m * HDIM + n];
            hnew[(size_t)m * HDIM + n] = (1.0f - z) * nn + z * h0;
        }
    }
}

// ---------------------------------------------------------------------------
// Kernel 3: per-row exp_map_zero of h_new -> fp32 output
// ---------------------------------------------------------------------------
__global__ __launch_bounds__(256) void row_exp_map_kernel(
    const float* __restrict__ hnew, float* __restrict__ out)
{
    __shared__ float red[256];
    const int row = blockIdx.x;
    const int t   = threadIdx.x;
    const size_t base = (size_t)row * HDIM;

    float v0 = hnew[base + t], v1 = hnew[base + t + 256];
    red[t] = v0 * v0 + v1 * v1;
    __syncthreads();
    for (int s = 128; s > 0; s >>= 1) {
        if (t < s) red[t] += red[t + s];
        __syncthreads();
    }
    float nm = fmaxf(sqrtf(red[0]), 1e-15f);
    float sc = fast_tanh(SQRT_C * nm) / (SQRT_C * nm);

    out[base + t]       = sc * v0;
    out[base + t + 256] = sc * v1;
}

// ---------------------------------------------------------------------------
// Host launcher
// ---------------------------------------------------------------------------
extern "C" void kernel_launch(void* const* d_in, const int* in_sizes, int n_in,
                              void* d_out, int out_size, void* d_ws, size_t ws_size,
                              hipStream_t stream)
{
    (void)in_sizes; (void)n_in; (void)out_size; (void)ws_size;

    const float* x   = (const float*)d_in[0];
    const float* h   = (const float*)d_in[1];
    const float* wih = (const float*)d_in[2];
    const float* whh = (const float*)d_in[3];
    const float* bih = (const float*)d_in[4];
    const float* bhh = (const float*)d_in[5];
    float* out = (float*)d_out;

    char* ws = (char*)d_ws;
    const size_t XT16_BYTES = (size_t)B_ROWS * HDIM * 2;   // 32 MB
    const size_t W16_BYTES  = (size_t)GDIM * HDIM * 2;     // 1.5 MB
    const size_t F32_BYTES  = (size_t)B_ROWS * HDIM * 4;   // 64 MB

    _Float16* xt16  = (_Float16*)(ws);
    _Float16* ht16  = (_Float16*)(ws + XT16_BYTES);
    _Float16* wih16 = (_Float16*)(ws + 2 * XT16_BYTES);
    _Float16* whh16 = (_Float16*)(ws + 2 * XT16_BYTES + W16_BYTES);
    float*    htf32 = (float*)   (ws + 2 * XT16_BYTES + 2 * W16_BYTES);
    float*    hnew  = (float*)   (ws + 2 * XT16_BYTES + 2 * W16_BYTES + F32_BYTES);

    {
        int nblk = (GDIM * HDIM + 255) / 256;
        cvt_weights_kernel<<<nblk, 256, 0, stream>>>(wih, whh, wih16, whh16);
    }
    row_log_map_kernel<<<B_ROWS, 256, 0, stream>>>(x, h, xt16, ht16, htf32);
    {
        dim3 grid(HDIM / 64, B_ROWS / M_BLK);   // 8 x 512
        gru_gemm_kernel<<<grid, 256, 0, stream>>>(xt16, ht16, wih16, whh16,
                                                  bih, bhh, htf32, hnew);
    }
    row_exp_map_kernel<<<B_ROWS, 256, 0, stream>>>(hnew, out);
}
